// FinancialContagionGNN_45603962749680
// MI455X (gfx1250) — matrix operand order repaired and checked by reading, not compiled, not run on hardware
//
#include <hip/hip_runtime.h>
#include <hip/hip_bf16.h>
#include <math.h>

// Problem constants (match reference).
#define N_NODES 30000
#define N_EDGES 480000
#define NFEAT   128
#define EFEAT   64
#define HID     256
#define NLAYERS 4
#define NHEADS  8
#define DHEAD   32
#define HID2    128   // HID/2, MLP hidden

typedef __attribute__((ext_vector_type(16))) _Float16 v16h;
typedef __attribute__((ext_vector_type(8)))  _Float16 v8h;
typedef __attribute__((ext_vector_type(8)))  float    v8f;

// ---------------------------------------------------------------------------
// float atomic max via CAS (scatter-softmax max)
// ---------------------------------------------------------------------------
__device__ __forceinline__ void atomicMaxF(float* addr, float val) {
  int* ia = (int*)addr;
  int old = __float_as_int(*addr);
  while (val > __int_as_float(old)) {
    int assumed = old;
    old = atomicCAS(ia, assumed, __float_as_int(val));
    if (old == assumed) break;
  }
}

// ---------------------------------------------------------------------------
// WMMA GEMM:  C[M,NOUT] = A[M,K] @ B[K,NOUT] (+bias) (+relu)
// f32 inputs staged through LDS as f16; f32 WMMA accumulators.
// K, NOUT compile-time (K % 32 == 0, NOUT in {128,256}); M % 16 == 0.
// Block = 256 threads (8 waves). Block owns rows [16*bid, 16*bid+16);
// wave w owns column tiles {w} (NOUT=128) or {w, w+8} (NOUT=256).
// LDS layout gives each lane contiguous 16B fragment chunks -> ds_load_b128.
//   Ah: row-major [16][k], stride 40 halves (80B, 16B-aligned chunks)
//   Bt: TRANSPOSED [NOUT cols][k], stride 40 halves
// ---------------------------------------------------------------------------
template <int K, int NOUT>
__global__ void gemm_wmma_f16(const float* __restrict__ A,
                              const float* __restrict__ B,
                              const float* __restrict__ bias,
                              float* __restrict__ C,
                              int relu) {
  constexpr int AS = 40;                  // LDS stride in halves
  __shared__ _Float16 Ah[16 * AS];
  __shared__ _Float16 Bt[NOUT * AS];

  const int tid  = threadIdx.x;
  const int lane = tid & 31;
  const int wv   = tid >> 5;              // wave id 0..7
  const int lrow = lane & 15;
  const int hsel = lane >> 4;             // lane half: 0 or 1
  const int row0 = blockIdx.x * 16;
  constexpr int numT = NOUT >> 7;         // 128 -> 1 tile/wave, 256 -> 2

  v8f acc[2] = {};

#pragma unroll
  for (int k0 = 0; k0 < K; k0 += 32) {
    __syncthreads();
    // Stage A tile: 16 rows x 32 k (f32 -> f16)
#pragma unroll
    for (int i = tid; i < 16 * 32; i += 256) {
      int r = i >> 5, kk = i & 31;
      Ah[r * AS + kk] = (_Float16)A[(size_t)(row0 + r) * K + (k0 + kk)];
    }
    // Stage B panel transposed: Bt[n][kk] = B[k0+kk][n] (f32 -> f16)
#pragma unroll
    for (int i = tid; i < 32 * NOUT; i += 256) {
      int kk = i / NOUT, n = i - kk * NOUT;
      Bt[n * AS + kk] = (_Float16)B[(size_t)(k0 + kk) * NOUT + n];
    }
    // Prefetch next K-panel into cache while we compute (global_prefetch_b8)
    if (k0 + 32 < K) {
      __builtin_prefetch(&A[(size_t)(row0 + (tid >> 4)) * K + (k0 + 32)], 0, 1);
      __builtin_prefetch(&B[(size_t)(k0 + 32 + (tid >> 3)) * NOUT + ((tid & 7) << 5)], 0, 1);
    }
    __syncthreads();

    // A fragment (ISA 16-bit A 16x32 layout): per lane, row = lrow,
    //   elems 0..7  -> K = hsel*8 + 0..7      (one 16B LDS chunk)
    //   elems 8..15 -> K = 16 + hsel*8 + 0..7 (one 16B LDS chunk)
    const v8h* ap = (const v8h*)(Ah + lrow * AS);
    v8h a_lo = ap[hsel];
    v8h a_hi = ap[2 + hsel];
    v16h af = __builtin_shufflevector(a_lo, a_hi,
        0, 1, 2, 3, 4, 5, 6, 7, 8, 9, 10, 11, 12, 13, 14, 15);

#pragma unroll
    for (int t = 0; t < numT; ++t) {      // compile-time; EXEC stays all-ones
      int ct = wv + t * 8;
      // B fragment (16-bit B 32x16): col = ct*16+lrow, K = hsel*8 + 0..7, 16 + hsel*8 + 0..7
      const v8h* bp = (const v8h*)(Bt + (ct * 16 + lrow) * AS);
      v8h b_lo = bp[hsel];
      v8h b_hi = bp[2 + hsel];
      v16h bf = __builtin_shufflevector(b_lo, b_hi,
          0, 1, 2, 3, 4, 5, 6, 7, 8, 9, 10, 11, 12, 13, 14, 15);
      acc[t] = __builtin_amdgcn_wmma_f32_16x16x32_f16(
          false, af, false, bf, (short)0, acc[t], false, false);
    }
  }

  // C/D layout: elem r -> row = row0 + 8*hsel + r, col = ct*16 + lrow
#pragma unroll
  for (int t = 0; t < numT; ++t) {
    int ct   = wv + t * 8;
    int ncol = ct * 16 + lrow;
    float bb = bias ? bias[ncol] : 0.0f;
#pragma unroll
    for (int r = 0; r < 8; ++r) {
      int row   = row0 + hsel * 8 + r;
      float val = acc[t][r] + bb;
      if (relu) val = fmaxf(val, 0.0f);
      C[(size_t)row * NOUT + ncol] = val;
    }
  }
}

// ---------------------------------------------------------------------------
// Utility fill
// ---------------------------------------------------------------------------
__global__ void fill_kernel(float* __restrict__ p, float v, int n) {
  int i = blockIdx.x * blockDim.x + threadIdx.x;
  if (i < n) p[i] = v;
}

// ---------------------------------------------------------------------------
// Edge attention kernels
// ---------------------------------------------------------------------------
__global__ void edge_score_kernel(const float* __restrict__ q,
                                  const float* __restrict__ k,
                                  const float* __restrict__ ke,
                                  const int* __restrict__ src,
                                  const int* __restrict__ dst,
                                  float* __restrict__ score,
                                  float* __restrict__ smax) {
  int t = blockIdx.x * blockDim.x + threadIdx.x;
  if (t >= N_EDGES * NHEADS) return;
  int e = t >> 3, hh = t & 7;
  int s = src[e], d = dst[e];
  const float* qp  = q  + (size_t)d * HID + hh * DHEAD;
  const float* kp  = k  + (size_t)s * HID + hh * DHEAD;
  const float* kep = ke + (size_t)e * HID + hh * DHEAD;
  float acc = 0.f;
#pragma unroll 8
  for (int i = 0; i < DHEAD; ++i) acc += qp[i] * (kp[i] + kep[i]);
  acc *= 0.17677669529663687f;  // 1/sqrt(32)
  score[t] = acc;
  atomicMaxF(&smax[(size_t)d * NHEADS + hh], acc);
}

__global__ void edge_p_kernel(const float* __restrict__ score,
                              const float* __restrict__ smax,
                              const int* __restrict__ dst,
                              float* __restrict__ p,
                              float* __restrict__ denom) {
  int t = blockIdx.x * blockDim.x + threadIdx.x;
  if (t >= N_EDGES * NHEADS) return;
  int e = t >> 3, hh = t & 7;
  int d = dst[e];
  float pm = __expf(score[t] - smax[(size_t)d * NHEADS + hh]);
  p[t] = pm;
  atomicAdd(&denom[(size_t)d * NHEADS + hh], pm);
}

// one block per edge, one thread per channel
__global__ void edge_agg_kernel(const float* __restrict__ p,
                                const float* __restrict__ denom,
                                const float* __restrict__ v,
                                const float* __restrict__ ke,
                                const int* __restrict__ src,
                                const int* __restrict__ dst,
                                float* __restrict__ agg) {
  int e = blockIdx.x;
  int c = threadIdx.x;
  int hh = c >> 5;
  int s = src[e], d = dst[e];
  float alpha = p[(size_t)e * NHEADS + hh] /
                (denom[(size_t)d * NHEADS + hh] + 1e-16f);
  float val = alpha * (v[(size_t)s * HID + c] + ke[(size_t)e * HID + c]);
  atomicAdd(&agg[(size_t)d * HID + c], val);
}

// ---------------------------------------------------------------------------
// Residual + LayerNorm (block per node, 256 threads)
// h[n] = LN(h[n] + agg[n] + sk[n])
// ---------------------------------------------------------------------------
__global__ void node_update_ln(float* __restrict__ h,
                               const float* __restrict__ agg,
                               const float* __restrict__ sk,
                               const float* __restrict__ lng,
                               const float* __restrict__ lnb) {
  __shared__ float red[256];
  int n = blockIdx.x, c = threadIdx.x;
  size_t idx = (size_t)n * HID + c;
  float val = h[idx] + agg[idx] + sk[idx];
  red[c] = val;
  __syncthreads();
  for (int s = 128; s > 0; s >>= 1) {
    if (c < s) red[c] += red[c + s];
    __syncthreads();
  }
  float m = red[0] * (1.0f / HID);
  __syncthreads();
  float dv = val - m;
  red[c] = dv * dv;
  __syncthreads();
  for (int s = 128; s > 0; s >>= 1) {
    if (c < s) red[c] += red[c + s];
    __syncthreads();
  }
  float var = red[0] * (1.0f / HID);
  h[idx] = dv * rsqrtf(var + 1e-5f) * lng[c] + lnb[c];
}

// ---------------------------------------------------------------------------
// Global mean pool (block per channel)
// ---------------------------------------------------------------------------
__global__ void pool_mean(const float* __restrict__ h, float* __restrict__ g) {
  __shared__ float red[256];
  int c = blockIdx.x;
  float s = 0.f;
  for (int n = threadIdx.x; n < N_NODES; n += 256)
    s += h[(size_t)n * HID + c];
  red[threadIdx.x] = s;
  __syncthreads();
  for (int st = 128; st > 0; st >>= 1) {
    if (threadIdx.x < st) red[threadIdx.x] += red[threadIdx.x + st];
    __syncthreads();
  }
  if (threadIdx.x == 0) g[c] = red[0] / (float)N_NODES;
}

// ---------------------------------------------------------------------------
// MLP head second stages
// ---------------------------------------------------------------------------
__global__ void contagion_out(const float* __restrict__ hid,
                              const float* __restrict__ W2,
                              const float* __restrict__ b2,
                              float* __restrict__ out) {
  int n = blockIdx.x * blockDim.x + threadIdx.x;
  if (n >= N_NODES) return;
  const float* hp = hid + (size_t)n * HID2;
  float z = b2[0];
#pragma unroll 8
  for (int j = 0; j < HID2; ++j) z += hp[j] * W2[j];
  out[n] = 1.0f / (1.0f + __expf(-z));
}

__global__ void systemic_head(const float* __restrict__ g,
                              const float* __restrict__ W1,
                              const float* __restrict__ b1,
                              const float* __restrict__ W2,
                              const float* __restrict__ b2,
                              float* __restrict__ out) {
  __shared__ float hred[HID2];
  int j = threadIdx.x;  // 0..127
  float a = b1[j];
  for (int c = 0; c < HID; ++c) a += g[c] * W1[c * HID2 + j];
  hred[j] = fmaxf(a, 0.0f);
  __syncthreads();
  if (j == 0) {
    float z = b2[0];
    for (int jj = 0; jj < HID2; ++jj) z += hred[jj] * W2[jj];
    out[0] = 1.0f / (1.0f + __expf(-z));
  }
}

// ---------------------------------------------------------------------------
// Host launch
// ---------------------------------------------------------------------------
extern "C" void kernel_launch(void* const* d_in, const int* in_sizes, int n_in,
                              void* d_out, int out_size, void* d_ws, size_t ws_size,
                              hipStream_t stream) {
  (void)in_sizes; (void)n_in; (void)out_size; (void)ws_size;

  const float* x         = (const float*)d_in[0];
  const float* edge_attr = (const float*)d_in[1];
  const int*   edge_idx  = (const int*)d_in[2];
  const int*   src       = edge_idx;
  const int*   dst       = edge_idx + N_EDGES;
  const float* enc_Wn = (const float*)d_in[3];
  const float* enc_bn = (const float*)d_in[4];
  const float* enc_We = (const float*)d_in[5];
  const float* enc_be = (const float*)d_in[6];
  const float* Wq  = (const float*)d_in[7];
  const float* bq  = (const float*)d_in[8];
  const float* Wk  = (const float*)d_in[9];
  const float* bk  = (const float*)d_in[10];
  const float* Wv  = (const float*)d_in[11];
  const float* bv  = (const float*)d_in[12];
  const float* We  = (const float*)d_in[13];
  const float* Wsk = (const float*)d_in[14];
  const float* bsk = (const float*)d_in[15];
  const float* ln_g = (const float*)d_in[16];
  const float* ln_b = (const float*)d_in[17];
  const float* cp_W1 = (const float*)d_in[18];
  const float* cp_b1 = (const float*)d_in[19];
  const float* cp_W2 = (const float*)d_in[20];
  const float* cp_b2 = (const float*)d_in[21];
  const float* sp_W1 = (const float*)d_in[22];
  const float* sp_b1 = (const float*)d_in[23];
  const float* sp_W2 = (const float*)d_in[24];
  const float* sp_b2 = (const float*)d_in[25];

  // Output layout: h [N,HID], g [1,HID], contagion [N,1], systemic [1,1]
  float* out       = (float*)d_out;
  float* h         = out;
  float* g         = out + (size_t)N_NODES * HID;
  float* contagion = g + HID;
  float* systemic  = contagion + N_NODES;

  // Workspace layout (floats). Total ~296M floats (~1.2 GB).
  float* ws = (float*)d_ws;
  size_t off = 0;
  float* e_enc = ws + off; off += (size_t)N_EDGES * HID;   // [E,HID]
  float* ke    = ws + off; off += (size_t)N_EDGES * HID;   // [E,HID]
  float* q     = ws + off; off += (size_t)N_NODES * HID;
  float* k     = ws + off; off += (size_t)N_NODES * HID;
  float* v     = ws + off; off += (size_t)N_NODES * HID;
  float* sk    = ws + off; off += (size_t)N_NODES * HID;
  float* agg   = ws + off; off += (size_t)N_NODES * HID;
  float* score = ws + off; off += (size_t)N_EDGES * NHEADS;
  float* pbuf  = ws + off; off += (size_t)N_EDGES * NHEADS;
  float* smax  = ws + off; off += (size_t)N_NODES * NHEADS;
  float* denom = ws + off; off += (size_t)N_NODES * NHEADS;
  float* hid1  = ws + off; off += (size_t)N_NODES * HID2;

  const int TB = 256;
  const int gridN16 = N_NODES / 16;   // 1875
  const int gridE16 = N_EDGES / 16;   // 30000
  const int nhTot   = N_NODES * NHEADS;
  const int ehTot   = N_EDGES * NHEADS;

  // Encoders
  gemm_wmma_f16<NFEAT, HID><<<gridN16, TB, 0, stream>>>(x, enc_Wn, enc_bn, h, 0);
  gemm_wmma_f16<EFEAT, HID><<<gridE16, TB, 0, stream>>>(edge_attr, enc_We, enc_be, e_enc, 0);

  for (int l = 0; l < NLAYERS; ++l) {
    const float* Wq_l  = Wq  + (size_t)l * HID * HID;
    const float* Wk_l  = Wk  + (size_t)l * HID * HID;
    const float* Wv_l  = Wv  + (size_t)l * HID * HID;
    const float* We_l  = We  + (size_t)l * HID * HID;
    const float* Wsk_l = Wsk + (size_t)l * HID * HID;
    const float* bq_l  = bq  + (size_t)l * HID;
    const float* bk_l  = bk  + (size_t)l * HID;
    const float* bv_l  = bv  + (size_t)l * HID;
    const float* bsk_l = bsk + (size_t)l * HID;

    // Dense projections (WMMA)
    gemm_wmma_f16<HID, HID><<<gridN16, TB, 0, stream>>>(h, Wq_l,  bq_l,  q,  0);
    gemm_wmma_f16<HID, HID><<<gridN16, TB, 0, stream>>>(h, Wk_l,  bk_l,  k,  0);
    gemm_wmma_f16<HID, HID><<<gridN16, TB, 0, stream>>>(h, Wv_l,  bv_l,  v,  0);
    gemm_wmma_f16<HID, HID><<<gridN16, TB, 0, stream>>>(h, Wsk_l, bsk_l, sk, 0);
    gemm_wmma_f16<HID, HID><<<gridE16, TB, 0, stream>>>(e_enc, We_l, nullptr, ke, 0);

    // Init scatter buffers
    fill_kernel<<<(nhTot + TB - 1) / TB, TB, 0, stream>>>(smax, -INFINITY, nhTot);
    fill_kernel<<<(nhTot + TB - 1) / TB, TB, 0, stream>>>(denom, 0.0f, nhTot);
    fill_kernel<<<(N_NODES * HID + TB - 1) / TB, TB, 0, stream>>>(agg, 0.0f, N_NODES * HID);

    // Scatter-softmax attention
    edge_score_kernel<<<(ehTot + TB - 1) / TB, TB, 0, stream>>>(q, k, ke, src, dst, score, smax);
    edge_p_kernel<<<(ehTot + TB - 1) / TB, TB, 0, stream>>>(score, smax, dst, pbuf, denom);
    edge_agg_kernel<<<N_EDGES, TB, 0, stream>>>(pbuf, denom, v, ke, src, dst, agg);

    // Residual + shared LayerNorm
    node_update_ln<<<N_NODES, TB, 0, stream>>>(h, agg, sk, ln_g, ln_b);
  }

  // Readout
  pool_mean<<<HID, TB, 0, stream>>>(h, g);
  gemm_wmma_f16<HID, HID2><<<gridN16, TB, 0, stream>>>(h, cp_W1, cp_b1, hid1, 1);
  contagion_out<<<(N_NODES + TB - 1) / TB, TB, 0, stream>>>(hid1, cp_W2, cp_b2, contagion);
  systemic_head<<<1, HID2, 0, stream>>>(g, sp_W1, sp_b1, sp_W2, sp_b2, systemic);
}
